// CA_36670430773307
// MI455X (gfx1250) — matrix operand order repaired and checked by reading, not compiled, not run on hardware
//
#include <hip/hip_runtime.h>
#include <hip/hip_bf16.h>

typedef __attribute__((ext_vector_type(4)))  _Float16 v4h;
typedef __attribute__((ext_vector_type(8)))  _Float16 v8h;
typedef __attribute__((ext_vector_type(16))) _Float16 v16h;
typedef __attribute__((ext_vector_type(4)))  float    v4f;
typedef __attribute__((ext_vector_type(8)))  float    v8f;

#define DIMC   512
#define NSEQ   2048
#define BATCH  4
#define HEADS  8
#define HDIM   64
#define ROWS   (BATCH * NSEQ)   /* 8192 */
#define QKVC   (3 * DIMC)       /* 1536 */
#define QSCALE 0.125f           /* (512/8)^-0.5 */
#define LN_EPS 1e-5f
#define BPAD   528               /* 512 + 16 halves: 32B-aligned rows, fewer bank conflicts */
#define KPAD   80                /* 64 + 16 halves */

// ---- WMMA fragment helpers (wave32, v_wmma_f32_16x16x32_f16) -------------
// A 16x32 f16: lane (hi=lane>>4, m=lane&15); elems 0..7 -> K=hi*8+e,
// elems 8..15 -> K=16+hi*8+(e-8).  Two contiguous 16B loads per lane.
static __device__ inline v16h load_a_frag(const _Float16* base, int ld, int kb, int lane) {
  int m = lane & 15, hi = lane >> 4;
  const _Float16* p = base + (size_t)m * ld + kb + hi * 8;
  v8h lo  = *(const v8h*)p;
  v8h hiv = *(const v8h*)(p + 16);
  return __builtin_shufflevector(lo, hiv, 0,1,2,3,4,5,6,7,8,9,10,11,12,13,14,15);
}
// B 32x16 f16 (stored as BT: N rows x K cols, row-major): lane (hi,n);
// elems 0..7 -> K=hi*8+e, elems 8..15 -> K=16+hi*8+(e-8).  Two 16B loads per lane.
static __device__ inline v16h load_b_frag(const _Float16* base, int ld, int kb, int lane) {
  int n = lane & 15, hi = lane >> 4;
  const _Float16* p = base + (size_t)n * ld + kb + hi * 8;
  v8h lo  = *(const v8h*)p;
  v8h hiv = *(const v8h*)(p + 16);
  return __builtin_shufflevector(lo, hiv, 0,1,2,3,4,5,6,7,8,9,10,11,12,13,14,15);
}
static __device__ inline v8f wmma_f16(v16h a, v16h b, v8f c) {
  return __builtin_amdgcn_wmma_f32_16x16x32_f16(false, a, false, b, (short)0, c, false, false);
}

// ---- 1) weight transpose + fp32->fp16 cast -------------------------------
__global__ void k_transpose_cast(const float* __restrict__ in, _Float16* __restrict__ out,
                                 int R, int C) {
  int idx = blockIdx.x * 256 + threadIdx.x;
  if (idx >= R * C) return;
  int r = idx / C, c = idx - r * C;
  out[(size_t)c * R + r] = (_Float16)in[idx];
}

// ---- 2) LayerNorm (one wave per 512-wide row) ----------------------------
__global__ void k_layernorm(const float* __restrict__ x, const float* __restrict__ g,
                            const float* __restrict__ bta, _Float16* __restrict__ out) {
  int row  = blockIdx.x * 4 + (threadIdx.x >> 5);
  int lane = threadIdx.x & 31;
  const float* xr = x + (size_t)row * DIMC;
  v4f v[4];
  float s = 0.f, ss = 0.f;
#pragma unroll
  for (int j = 0; j < 4; ++j) {
    v[j] = *(const v4f*)(xr + lane * 4 + j * 128);
#pragma unroll
    for (int e = 0; e < 4; ++e) { s += v[j][e]; ss += v[j][e] * v[j][e]; }
  }
#pragma unroll
  for (int off = 1; off < 32; off <<= 1) {
    s  += __shfl_xor(s,  off, 32);
    ss += __shfl_xor(ss, off, 32);
  }
  float mu  = s * (1.f / DIMC);
  float var = ss * (1.f / DIMC) - mu * mu;
  float rs  = rsqrtf(var + LN_EPS);
  _Float16* orow = out + (size_t)row * DIMC;
#pragma unroll
  for (int j = 0; j < 4; ++j) {
    v4h h;
#pragma unroll
    for (int e = 0; e < 4; ++e) {
      int col = lane * 4 + j * 128 + e;
      h[e] = (_Float16)((v[j][e] - mu) * rs * g[col] + bta[col]);
    }
    *(v4h*)(orow + lane * 4 + j * 128) = h;
  }
}

// ---- GEMM core: block stages 64-col B panel in LDS; wave = 16x64 tile ----
// Bs: [64][BPAD] LDS panel (staged by caller's block).  A double-buffered
// from global; B fragments from LDS (ds_load, low latency).
static __device__ inline void gemm_16x64_lds(const _Float16* __restrict__ A, int lda,
                                             const _Float16 (*Bs)[BPAD], int lane,
                                             v8f acc[4]) {
  v16h a = load_a_frag(A, lda, 0, lane);
#pragma unroll 4
  for (int kb = 0; kb < DIMC - 32; kb += 32) {
    v16h an = load_a_frag(A, lda, kb + 32, lane);   // next A from global
#pragma unroll
    for (int c = 0; c < 4; ++c)
      acc[c] = wmma_f16(a, load_b_frag(&Bs[c * 16][0], BPAD, kb, lane), acc[c]);
    a = an;
  }
#pragma unroll
  for (int c = 0; c < 4; ++c)
    acc[c] = wmma_f16(a, load_b_frag(&Bs[c * 16][0], BPAD, DIMC - 32, lane), acc[c]);
}

static __device__ inline void stage_b_panel(const _Float16* __restrict__ BT, int col0,
                                            _Float16 (*Bs)[BPAD]) {
  // 64 rows x 512 halves = 4096 16B-chunks; 256 threads x 16 chunks
#pragma unroll
  for (int it = 0; it < 16; ++it) {
    int chunk = it * 256 + threadIdx.x;
    int j = chunk >> 6, o = (chunk & 63) * 8;
    *(v8h*)&Bs[j][o] = *(const v8h*)(BT + (size_t)(col0 + j) * DIMC + o);
  }
  __syncthreads();
}

// ---- 3) QKV GEMM: [8192x512]@[512x1536]; Q pre-scaled; V also -> VT ------
__global__ void k_qkv_gemm(const _Float16* __restrict__ X, const _Float16* __restrict__ WT,
                           _Float16* __restrict__ QKV, _Float16* __restrict__ VT) {
  __shared__ __align__(32) _Float16 Bs[64][BPAD];
  int lane = threadIdx.x & 31;
  int wave = threadIdx.x >> 5;
  int row0 = blockIdx.y * 128 + wave * 16;
  int col0 = blockIdx.x * 64;
  stage_b_panel(WT, col0, Bs);
  v8f acc[4] = {};
  gemm_16x64_lds(X + (size_t)row0 * DIMC, DIMC, Bs, lane, acc);
  int hi = lane >> 4, n = lane & 15;
#pragma unroll
  for (int c = 0; c < 4; ++c) {
    int col = col0 + c * 16 + n;
    float scale = (col < DIMC) ? QSCALE : 1.f;
#pragma unroll
    for (int r = 0; r < 8; ++r) {
      int row = row0 + hi * 8 + r;
      float vv = acc[c][r] * scale;
      QKV[(size_t)row * QKVC + col] = (_Float16)vv;
      if (col >= 2 * DIMC) {
        int bb = row >> 11, nn = row & 2047;
        int hh = (col - 2 * DIMC) >> 6, dd = (col - 2 * DIMC) & 63;
        VT[((size_t)((bb * HEADS + hh) * HDIM + dd)) * NSEQ + nn] = (_Float16)vv;
      }
    }
  }
}

// ---- 4) Flash cross-attention; block stages K/V tiles in LDS -------------
__global__ void k_attention(const _Float16* __restrict__ QKV_i, const _Float16* __restrict__ QKV_t,
                            const _Float16* __restrict__ VT_i,  const _Float16* __restrict__ VT_t,
                            _Float16* __restrict__ att_i2t, _Float16* __restrict__ att_t2i) {
  int lane = threadIdx.x & 31;
  int wave = threadIdx.x >> 5;
  int h   = blockIdx.y;
  int dir = blockIdx.z >> 2;   // 0: i->t   1: t->i
  int b   = blockIdx.z & 3;
  const _Float16* Qsrc = dir ? QKV_t : QKV_i;
  const _Float16* Ksrc = dir ? QKV_i : QKV_t;
  const _Float16* Vt   = dir ? VT_i  : VT_t;
  _Float16*       outb = dir ? att_t2i : att_i2t;

  int qrow0 = b * NSEQ + blockIdx.x * 64 + wave * 16;
  const _Float16* Aq = Qsrc + (size_t)qrow0 * QKVC + h * HDIM;
  v16h q0 = load_a_frag(Aq, QKVC, 0,  lane);
  v16h q1 = load_a_frag(Aq, QKVC, 32, lane);
  const _Float16* Kbase0 = Ksrc + (size_t)(b * NSEQ) * QKVC + DIMC + h * HDIM;
  const _Float16* VTb    = Vt + (size_t)((b * HEADS + h) * HDIM) * NSEQ;

  float mst[8], lst[8];
  v8f acc[4] = {};
#pragma unroll
  for (int r = 0; r < 8; ++r) { mst[r] = -3.0e38f; lst[r] = 0.f; }

  __shared__ __align__(32) _Float16 Ks[64][KPAD];      // keys x dims
  __shared__ __align__(32) _Float16 Vs[64][KPAD];      // dims x keys (V^T)
  __shared__ __align__(32) _Float16 pb[4][16][KPAD];   // P tile per wave
  int hi = lane >> 4, n = lane & 15;

  for (int key0 = 0; key0 < NSEQ; key0 += 64) {
    // prefetch next key block into L2 (global_prefetch_b8)
    __builtin_prefetch(Kbase0 + (size_t)(key0 + 64 + lane) * QKVC, 0, 0);
    __builtin_prefetch(Kbase0 + (size_t)(key0 + 96 + lane) * QKVC, 0, 0);
    __builtin_prefetch(VTb + (size_t)lane * NSEQ + key0 + 64, 0, 0);
    __builtin_prefetch(VTb + (size_t)(32 + lane) * NSEQ + key0 + 64, 0, 0);

    // cooperative stage: K tile (64 keys x 64 d) + V^T tile (64 d x 64 keys)
#pragma unroll
    for (int it = 0; it < 4; ++it) {
      int chunk = it * 128 + threadIdx.x;             // 0..511
      int row = chunk >> 3, o = (chunk & 7) * 8;
      *(v8h*)&Ks[row][o] = *(const v8h*)(Kbase0 + (size_t)(key0 + row) * QKVC + o);
      *(v8h*)&Vs[row][o] = *(const v8h*)(VTb + (size_t)row * NSEQ + key0 + o);
    }
    __syncthreads();

    // S = Q K^T from LDS fragments (8 WMMAs)
    v16h kf[8];
#pragma unroll
    for (int t = 0; t < 4; ++t) {
      kf[2 * t]     = load_b_frag(&Ks[t * 16][0], KPAD, 0,  lane);
      kf[2 * t + 1] = load_b_frag(&Ks[t * 16][0], KPAD, 32, lane);
    }
    v8f s[4] = {};
#pragma unroll
    for (int t = 0; t < 4; ++t) {
      s[t] = wmma_f16(q0, kf[2 * t],     s[t]);
      s[t] = wmma_f16(q1, kf[2 * t + 1], s[t]);
    }

    // issue V fragment ds-loads now; softmax hides their latency
    v16h vf[8];
#pragma unroll
    for (int c = 0; c < 4; ++c) {
      vf[2 * c]     = load_b_frag(&Vs[c * 16][0], KPAD, 0,  lane);
      vf[2 * c + 1] = load_b_frag(&Vs[c * 16][0], KPAD, 32, lane);
    }

    // online softmax update over 64 keys
#pragma unroll
    for (int r = 0; r < 8; ++r) {
      float mx = fmaxf(fmaxf(s[0][r], s[1][r]), fmaxf(s[2][r], s[3][r]));
      mx = fmaxf(mx, __shfl_xor(mx, 1, 32));
      mx = fmaxf(mx, __shfl_xor(mx, 2, 32));
      mx = fmaxf(mx, __shfl_xor(mx, 4, 32));
      mx = fmaxf(mx, __shfl_xor(mx, 8, 32));          // row max over 16 lanes
      float mnew  = fmaxf(mst[r], mx);
      float alpha = __expf(mst[r] - mnew);
      float p[4], rs = 0.f;
#pragma unroll
      for (int t = 0; t < 4; ++t) { p[t] = __expf(s[t][r] - mnew); rs += p[t]; }
      rs += __shfl_xor(rs, 1, 32);
      rs += __shfl_xor(rs, 2, 32);
      rs += __shfl_xor(rs, 4, 32);
      rs += __shfl_xor(rs, 8, 32);
      lst[r] = lst[r] * alpha + rs;
      mst[r] = mnew;
#pragma unroll
      for (int c = 0; c < 4; ++c) acc[c][r] *= alpha;
#pragma unroll
      for (int t = 0; t < 4; ++t)
        pb[wave][hi * 8 + r][t * 16 + n] = (_Float16)p[t];  // D-layout -> row-major
    }
    // cross-lane LDS hazard: same-wave, lockstep -> dscnt wait suffices
    asm volatile("s_wait_dscnt 0" ::: "memory");

    // O += P V  (8 WMMAs, V fragments already resident)
    v16h pa0 = load_a_frag(&pb[wave][0][0], KPAD, 0,  lane);
    v16h pa1 = load_a_frag(&pb[wave][0][0], KPAD, 32, lane);
#pragma unroll
    for (int c = 0; c < 4; ++c) {
      acc[c] = wmma_f16(pa0, vf[2 * c],     acc[c]);
      acc[c] = wmma_f16(pa1, vf[2 * c + 1], acc[c]);
    }
    __syncthreads();   // all waves done with Ks/Vs before next stage
  }

#pragma unroll
  for (int r = 0; r < 8; ++r) {
    float inv = 1.f / lst[r];
    int row = qrow0 + hi * 8 + r;
    _Float16* orow = outb + (size_t)row * DIMC + h * HDIM;
#pragma unroll
    for (int c = 0; c < 4; ++c)
      orow[c * 16 + n] = (_Float16)(acc[c][r] * inv);
  }
}

// ---- 5) Output projection: [8192x512]@[512x512] -> f32 concat layout -----
__global__ void k_out_proj(const _Float16* __restrict__ att, const _Float16* __restrict__ WT,
                           float* __restrict__ out, int rowOffset) {
  __shared__ __align__(32) _Float16 Bs[64][BPAD];
  int lane = threadIdx.x & 31;
  int wave = threadIdx.x >> 5;
  int row0 = blockIdx.y * 128 + wave * 16;
  int col0 = blockIdx.x * 64;
  stage_b_panel(WT, col0, Bs);
  v8f acc[4] = {};
  gemm_16x64_lds(att + (size_t)row0 * DIMC, DIMC, Bs, lane, acc);
  int hi = lane >> 4, n = lane & 15;
#pragma unroll
  for (int r = 0; r < 8; ++r) {
    int grow = row0 + hi * 8 + r;
    int bb = grow >> 11, nn = grow & 2047;
    float* orow = out + ((size_t)bb * 2 * NSEQ + rowOffset + nn) * DIMC;
#pragma unroll
    for (int c = 0; c < 4; ++c)
      orow[col0 + c * 16 + n] = acc[c][r];
  }
}

extern "C" void kernel_launch(void* const* d_in, const int* in_sizes, int n_in,
                              void* d_out, int out_size, void* d_ws, size_t ws_size,
                              hipStream_t stream) {
  (void)in_sizes; (void)n_in; (void)out_size; (void)ws_size;
  const float* in_i   = (const float*)d_in[0];
  const float* in_t   = (const float*)d_in[1];
  const float* g_i    = (const float*)d_in[2];
  const float* b_i    = (const float*)d_in[3];
  const float* g_t    = (const float*)d_in[4];
  const float* b_t    = (const float*)d_in[5];
  const float* wqkv_i = (const float*)d_in[6];
  const float* wqkv_t = (const float*)d_in[7];
  const float* wout_i = (const float*)d_in[8];
  const float* wout_t = (const float*)d_in[9];
  float* out = (float*)d_out;

  char* p = (char*)d_ws;
  auto carve = [&](size_t bytes) -> char* {
    char* r = p; p += (bytes + 255) & ~(size_t)255; return r;
  };
  _Float16* Xh_i  = (_Float16*)carve((size_t)ROWS * DIMC * 2);
  _Float16* Xh_t  = (_Float16*)carve((size_t)ROWS * DIMC * 2);
  _Float16* WTq_i = (_Float16*)carve((size_t)QKVC * DIMC * 2);
  _Float16* WTq_t = (_Float16*)carve((size_t)QKVC * DIMC * 2);
  _Float16* WTo_i = (_Float16*)carve((size_t)DIMC * DIMC * 2);
  _Float16* WTo_t = (_Float16*)carve((size_t)DIMC * DIMC * 2);
  _Float16* QKV_i = (_Float16*)carve((size_t)ROWS * QKVC * 2);
  _Float16* QKV_t = (_Float16*)carve((size_t)ROWS * QKVC * 2);
  _Float16* VT_i  = (_Float16*)carve((size_t)BATCH * HEADS * HDIM * NSEQ * 2);
  _Float16* VT_t  = (_Float16*)carve((size_t)BATCH * HEADS * HDIM * NSEQ * 2);
  _Float16* a_i2t = (_Float16*)carve((size_t)ROWS * DIMC * 2);
  _Float16* a_t2i = (_Float16*)carve((size_t)ROWS * DIMC * 2);

  // 1) weights -> fp16, transposed (BT = N x K row-major)
  k_transpose_cast<<<(DIMC * QKVC + 255) / 256, 256, 0, stream>>>(wqkv_i, WTq_i, DIMC, QKVC);
  k_transpose_cast<<<(DIMC * QKVC + 255) / 256, 256, 0, stream>>>(wqkv_t, WTq_t, DIMC, QKVC);
  k_transpose_cast<<<(DIMC * DIMC + 255) / 256, 256, 0, stream>>>(wout_i, WTo_i, DIMC, DIMC);
  k_transpose_cast<<<(DIMC * DIMC + 255) / 256, 256, 0, stream>>>(wout_t, WTo_t, DIMC, DIMC);
  // 2) layernorm -> fp16 activations
  k_layernorm<<<ROWS / 4, 128, 0, stream>>>(in_i, g_i, b_i, Xh_i);
  k_layernorm<<<ROWS / 4, 128, 0, stream>>>(in_t, g_t, b_t, Xh_t);
  // 3) QKV projections (WMMA, LDS-staged B panel)
  k_qkv_gemm<<<dim3(QKVC / 64, ROWS / 128), 256, 0, stream>>>(Xh_i, WTq_i, QKV_i, VT_i);
  k_qkv_gemm<<<dim3(QKVC / 64, ROWS / 128), 256, 0, stream>>>(Xh_t, WTq_t, QKV_t, VT_t);
  // 4) bidirectional flash cross-attention (WMMA, LDS-staged K/V tiles)
  k_attention<<<dim3(NSEQ / 64, HEADS, 2 * BATCH), 128, 0, stream>>>(QKV_i, QKV_t, VT_i, VT_t,
                                                                     a_i2t, a_t2i);
  // 5) output projections (WMMA): concat([t2i, i2t], axis=1)
  k_out_proj<<<dim3(DIMC / 64, ROWS / 128), 256, 0, stream>>>(a_t2i, WTo_i, out, 0);
  k_out_proj<<<dim3(DIMC / 64, ROWS / 128), 256, 0, stream>>>(a_i2t, WTo_t, out, NSEQ);
}
